// GRUDEncoder_38457137168736
// MI455X (gfx1250) — matrix operand order repaired and checked by reading, not compiled, not run on hardware
//
#include <hip/hip_runtime.h>
#include <hip/hip_bf16.h>
#include <math.h>

// ---------------------------------------------------------------------------
// GRU-D encoder for MI455X (gfx1250, wave32, WMMA f16 16x16x32).
// B=256, S=256, D=128, H=512.
// ---------------------------------------------------------------------------

constexpr int Bn = 256;
constexpr int Sn = 256;
constexpr int Dn = 128;
constexpr int Hn = 512;
constexpr int Mn = Sn * Bn;     // 65536 GEMM rows (time-major, b fastest)

typedef __attribute__((ext_vector_type(16))) _Float16 v16h;
typedef __attribute__((ext_vector_type(8)))  _Float16 v8h;
typedef __attribute__((ext_vector_type(8)))  float    v8f;
typedef int v4i __attribute__((vector_size(16)));

#ifndef __has_builtin
#define __has_builtin(x) 0
#endif
#if __has_builtin(__builtin_amdgcn_global_load_async_to_lds_b128) && \
    __has_builtin(__builtin_amdgcn_s_wait_asynccnt)
#define USE_ASYNC_LDS 1
#else
#define USE_ASYNC_LDS 0
#endif

static __device__ inline v8f zero8() {
  v8f z = {0.f, 0.f, 0.f, 0.f, 0.f, 0.f, 0.f, 0.f};
  return z;
}

// Async (ASYNCcnt-tracked) 16-byte global->LDS stage; falls back to a
// synchronous VGPR bounce if the builtin is unavailable on this toolchain.
// Builtin prototype (from hipcc diagnostic): (int4 AS1*, int4 AS3*, imm, imm).
static __device__ inline void stage16(const _Float16* g, _Float16* l) {
#if USE_ASYNC_LDS
  __builtin_amdgcn_global_load_async_to_lds_b128(
      (__attribute__((address_space(1))) v4i*)(unsigned long long)g,
      (__attribute__((address_space(3))) v4i*)(unsigned)(unsigned long long)l,
      0, 0);
#else
  *(v8h*)l = *(const v8h*)g;
#endif
}
static __device__ inline void stage_wait() {
#if USE_ASYNC_LDS
  __builtin_amdgcn_s_wait_asynccnt(0);
#endif
}

// A-fragment (16x32 f16, row-major source with leading dim ld):
// lane L (0-15): row = row0 + L, K-chunks [0..7],[16..23]; lanes 16-31: [8..15],[24..31].
static __device__ inline v16h load_a_frag(const _Float16* __restrict__ base, int ld,
                                          int row0, int k0, int lane) {
  const int r  = row0 + (lane & 15);
  const int hi = lane >> 4;
  const _Float16* p = base + r * ld + k0 + hi * 8;
  v8h lo = *(const v8h*)p;
  v8h hh = *(const v8h*)(p + 16);
  return __builtin_shufflevector(lo, hh, 0,1,2,3,4,5,6,7,8,9,10,11,12,13,14,15);
}

// Same but for an LDS-resident tile (rows 0..15, leading dim ld f16).
static __device__ inline v16h load_a_lds(const _Float16* base, int ld, int k0, int lane) {
  const int r  = lane & 15;
  const int hi = lane >> 4;
  const _Float16* p = base + r * ld + k0 + hi * 8;
  v8h lo = *(const v8h*)p;
  v8h hh = *(const v8h*)(p + 16);
  return __builtin_shufflevector(lo, hh, 0,1,2,3,4,5,6,7,8,9,10,11,12,13,14,15);
}

// B-fragment (32x16 f16) from pre-transposed weight Wt stored (N,K) row-major.
// lane L: column n = n0 + L%16; K-chunks [0..7],[16..23] (L<16) or [8..15],[24..31] (L>=16).
static __device__ inline v16h load_b_frag(const _Float16* __restrict__ wt, int ld,
                                          int n0, int k0, int lane) {
  const int n = n0 + (lane & 15);
  const _Float16* p = wt + n * ld + k0 + ((lane >> 4) << 3);
  v8h lo = *(const v8h*)p;
  v8h hh = *(const v8h*)(p + 16);
  return __builtin_shufflevector(lo, hh, 0,1,2,3,4,5,6,7,8,9,10,11,12,13,14,15);
}

static __device__ inline v8f wmma_f16(v16h a, v16h b, v8f c) {
  return __builtin_amdgcn_wmma_f32_16x16x32_f16(false, a, false, b, (short)0, c,
                                                false, false);
}

// ---------------------------------------------------------------------------
// K0: transpose+downconvert weight (K,N) f32 -> (N,K) f16.
// ---------------------------------------------------------------------------
__global__ void pack_w(const float* __restrict__ src, _Float16* __restrict__ dst,
                       int K, int N) {
  int i = blockIdx.x * blockDim.x + threadIdx.x;
  if (i >= K * N) return;
  int n = i / K, k = i % K;
  dst[i] = (_Float16)src[k * N + n];
}

// ---------------------------------------------------------------------------
// K1: per-feature mean of observed values. One block per feature d.
// ---------------------------------------------------------------------------
__global__ void mean_k(const float* __restrict__ x, const float* __restrict__ mask,
                       float* __restrict__ x_mean) {
  const int d = blockIdx.x, tid = threadIdx.x;
  float s = 0.f, c = 0.f;
  for (int i = tid; i < Mn; i += 256) {
    float m = mask[i * Dn + d];
    s += x[i * Dn + d] * m;
    c += m;
  }
  __shared__ float ss[256], cc[256];
  ss[tid] = s; cc[tid] = c; __syncthreads();
  for (int o = 128; o > 0; o >>= 1) {
    if (tid < o) { ss[tid] += ss[tid + o]; cc[tid] += cc[tid + o]; }
    __syncthreads();
  }
  if (tid == 0) x_mean[d] = ss[0] / fmaxf(cc[0], 1.0f);
}

// ---------------------------------------------------------------------------
// K2: per-sequence forward scan producing x_hat, mask, delta as f16 in
// time-major (S*B, D) layout. grid=B blocks, block=D threads.
// ---------------------------------------------------------------------------
__global__ void scan_k(const float* __restrict__ x, const float* __restrict__ t,
                       const float* __restrict__ mask,
                       const float* __restrict__ w_gx, const float* __restrict__ b_gx,
                       const float* __restrict__ x_mean,
                       _Float16* __restrict__ Xh, _Float16* __restrict__ Mh,
                       _Float16* __restrict__ Dl) {
  const int b = blockIdx.x, d = threadIdx.x;
  const float wg = w_gx[d], bg = b_gx[d], xm = x_mean[d];
  float delta = 0.f, m_prev = 0.f, xl = xm, x_prev = 0.f, t_prev = 0.f;
  for (int s = 0; s < Sn; ++s) {
    const float ts = t[b * Sn + s];
    const float dt = (s == 0) ? 0.f : (ts - t_prev);
    t_prev = ts;
    delta = dt + (1.f - m_prev) * delta;
    xl    = m_prev * x_prev + (1.f - m_prev) * xl;
    const float ms = mask[(b * Sn + s) * Dn + d];
    const float xs = x[(b * Sn + s) * Dn + d];
    const float gx = __expf(-fmaxf(wg * delta + bg, 0.f));
    const float xh = ms * xs + (1.f - ms) * (gx * xl + (1.f - gx) * xm);
    const int row = s * Bn + b;
    Xh[row * Dn + d] = (_Float16)xh;
    Mh[row * Dn + d] = (_Float16)ms;
    Dl[row * Dn + d] = (_Float16)delta;
    m_prev = ms; x_prev = xs;
  }
}

// ---------------------------------------------------------------------------
// K3: fused input projections.  blockIdx.y: sel = y>>1 (0:az 1:ar 2:ah 3:gamma_h),
// n-half = y&1. The block's 16x128 A tile(s) are staged once into LDS through
// the async global->LDS path (ASYNCcnt) instead of being re-read by all 8
// waves; each wave then computes a 16x32 output tile (2 WMMA n-tiles).
// ---------------------------------------------------------------------------
__global__ void __launch_bounds__(256)
gemm1(const _Float16* __restrict__ Xh, const _Float16* __restrict__ Mh,
      const _Float16* __restrict__ Dl,
      const _Float16* __restrict__ WzT, const _Float16* __restrict__ VzT,
      const _Float16* __restrict__ WrT, const _Float16* __restrict__ VrT,
      const _Float16* __restrict__ WhT, const _Float16* __restrict__ VhT,
      const _Float16* __restrict__ WghT,
      const float* __restrict__ bz, const float* __restrict__ br,
      const float* __restrict__ bh, const float* __restrict__ bgh,
      _Float16* __restrict__ AZ, _Float16* __restrict__ AR,
      _Float16* __restrict__ AH, _Float16* __restrict__ GH) {
  __shared__ __align__(16) _Float16 aT[2][16 * Dn];   // 2 x 4 KB A tiles
  const int m0    = blockIdx.x * 16;
  const int sel   = blockIdx.y >> 1;
  const int nhalf = blockIdx.y & 1;
  const int wave  = threadIdx.x >> 5;
  const int lane  = threadIdx.x & 31;
  const int n0    = nhalf * 256 + wave * 32;

  const _Float16 *A1, *A2 = nullptr, *B1, *B2 = nullptr;
  const float* bias;
  _Float16* out;
  if (sel == 0)      { A1 = Xh; B1 = WzT; A2 = Mh; B2 = VzT; bias = bz;  out = AZ; }
  else if (sel == 1) { A1 = Xh; B1 = WrT; A2 = Mh; B2 = VrT; bias = br;  out = AR; }
  else if (sel == 2) { A1 = Xh; B1 = WhT; A2 = Mh; B2 = VhT; bias = bh;  out = AH; }
  else               { A1 = Dl; B1 = WghT;                   bias = bgh; out = GH; }

  // Stage A tiles: 256 threads x one 16B chunk each per tile.
  {
    const int r = threadIdx.x >> 4, c = threadIdx.x & 15;   // 16 rows x 16 chunks
    stage16(A1 + (m0 + r) * Dn + c * 8, &aT[0][r * Dn + c * 8]);
    if (sel != 3)
      stage16(A2 + (m0 + r) * Dn + c * 8, &aT[1][r * Dn + c * 8]);
  }
  stage_wait();
  __syncthreads();

  v8f c0 = zero8(), c1 = zero8();
#pragma unroll
  for (int kt = 0; kt < 4; ++kt) {
    v16h a  = load_a_lds(aT[0], Dn, kt * 32, lane);
    v16h b0 = load_b_frag(B1, Dn, n0,      kt * 32, lane);
    v16h b1 = load_b_frag(B1, Dn, n0 + 16, kt * 32, lane);
    c0 = wmma_f16(a, b0, c0);
    c1 = wmma_f16(a, b1, c1);
  }
  if (sel != 3) {
#pragma unroll
    for (int kt = 0; kt < 4; ++kt) {
      v16h a  = load_a_lds(aT[1], Dn, kt * 32, lane);
      v16h b0 = load_b_frag(B2, Dn, n0,      kt * 32, lane);
      v16h b1 = load_b_frag(B2, Dn, n0 + 16, kt * 32, lane);
      c0 = wmma_f16(a, b0, c0);
      c1 = wmma_f16(a, b1, c1);
    }
  }
  // epilogue: C/D layout -> lane holds col n0+..+lane%16, vgpr v holds row v+8*(lane/16)
  const int hi = lane >> 4, nn = lane & 15;
#pragma unroll
  for (int j = 0; j < 2; ++j) {
    const int n = n0 + j * 16 + nn;
    const float bv = bias[n];
    v8f& c = j ? c1 : c0;
#pragma unroll
    for (int v = 0; v < 8; ++v) {
      const int m = m0 + v + 8 * hi;
      float val = c[v] + bv;
      if (sel == 3) val = __expf(-fmaxf(val, 0.f));
      out[m * Hn + n] = (_Float16)val;
    }
  }
}

// ---------------------------------------------------------------------------
// K4: persistent GRU recurrence. 16 workgroups, each owns 16 batch rows for all
// 256 steps (h@U is row-independent -> no cross-workgroup sync). h lives in LDS
// as f16; 8 waves each own a 64-column slice of H. U matrices stream from L2
// (1.5 MB packed f16, far below the 192 MB L2). Next-step gate rows are
// prefetched per-lane while the current step's WMMAs run.
// ---------------------------------------------------------------------------
__global__ void __launch_bounds__(256, 1)
recur(const _Float16* __restrict__ AZ, const _Float16* __restrict__ AR,
      const _Float16* __restrict__ AH, const _Float16* __restrict__ GH,
      const _Float16* __restrict__ UzT, const _Float16* __restrict__ UrT,
      const _Float16* __restrict__ UhT, _Float16* __restrict__ HT) {
  __shared__ __align__(16) _Float16 hL[16 * Hn];
  __shared__ __align__(16) _Float16 rhL[16 * Hn];
  const int bg   = blockIdx.x;          // batch group: rows bg*16 .. bg*16+15
  const int wave = threadIdx.x >> 5;
  const int lane = threadIdx.x & 31;
  const int nwb  = wave * 64;           // this wave's H-column slice base
  const int hi = lane >> 4, nn = lane & 15;

  for (int i = threadIdx.x; i < 16 * Hn; i += 256) hL[i] = (_Float16)0.f;
  __syncthreads();

  for (int s = 0; s < Sn; ++s) {
    const int rowbase = s * Bn + bg * 16;

    // (a) hidden-state decay: h *= gamma_h  (own column slice)
    for (int idx = lane; idx < 16 * 64; idx += 32) {
      const int r = idx >> 6, n = nwb + (idx & 63);
      float hv = (float)hL[r * Hn + n];
      float g  = (float)GH[(rowbase + r) * Hn + n];
      hL[r * Hn + n] = (_Float16)(hv * g);
    }
    if (s + 1 < Sn) {  // warm L2/L0 for next step's gate rows (per-lane rows)
      const int pr = rowbase + Bn + (lane & 15);
      __builtin_prefetch(AZ + pr * Hn + nwb, 0, 0);
      __builtin_prefetch(AR + pr * Hn + nwb, 0, 0);
      __builtin_prefetch(AH + pr * Hn + nwb, 0, 0);
      __builtin_prefetch(GH + pr * Hn + nwb, 0, 0);
    }
    __syncthreads();

    // (b) z and r gate GEMMs: 16x64 slice, K = 512
    v8f cz[4], cr[4];
#pragma unroll
    for (int j = 0; j < 4; ++j) { cz[j] = zero8(); cr[j] = zero8(); }
    for (int kt = 0; kt < 16; ++kt) {
      v16h a = load_a_lds(hL, Hn, kt * 32, lane);
#pragma unroll
      for (int j = 0; j < 4; ++j) {
        cz[j] = wmma_f16(a, load_b_frag(UzT, Hn, nwb + j * 16, kt * 32, lane), cz[j]);
        cr[j] = wmma_f16(a, load_b_frag(UrT, Hn, nwb + j * 16, kt * 32, lane), cr[j]);
      }
    }

    // (c) gates + write r*h into LDS
    float zg[32], hold[32];
#pragma unroll
    for (int j = 0; j < 4; ++j) {
      const int n = nwb + j * 16 + nn;
#pragma unroll
      for (int v = 0; v < 8; ++v) {
        const int r = v + 8 * hi;
        const int row = rowbase + r;
        const float az = (float)AZ[row * Hn + n];
        const float ar = (float)AR[row * Hn + n];
        const float z  = 1.f / (1.f + __expf(-(cz[j][v] + az)));
        const float rr = 1.f / (1.f + __expf(-(cr[j][v] + ar)));
        const float hv = (float)hL[r * Hn + n];
        zg[j * 8 + v] = z; hold[j * 8 + v] = hv;
        rhL[r * Hn + n] = (_Float16)(rr * hv);
      }
    }
    __syncthreads();

    // (d) candidate GEMM over (r*h), then blend and commit h_new
    v8f ch[4];
#pragma unroll
    for (int j = 0; j < 4; ++j) ch[j] = zero8();
    for (int kt = 0; kt < 16; ++kt) {
      v16h a = load_a_lds(rhL, Hn, kt * 32, lane);
#pragma unroll
      for (int j = 0; j < 4; ++j)
        ch[j] = wmma_f16(a, load_b_frag(UhT, Hn, nwb + j * 16, kt * 32, lane), ch[j]);
    }
#pragma unroll
    for (int j = 0; j < 4; ++j) {
      const int n = nwb + j * 16 + nn;
#pragma unroll
      for (int v = 0; v < 8; ++v) {
        const int r = v + 8 * hi;
        const int row = rowbase + r;
        const float ah = (float)AH[row * Hn + n];
        const float ht = tanhf(ch[j][v] + ah);
        const float z  = zg[j * 8 + v];
        const float hn = (1.f - z) * hold[j * 8 + v] + z * ht;
        hL[r * Hn + n] = (_Float16)hn;
        HT[row * Hn + n] = (_Float16)hn;
      }
    }
    __syncthreads();
  }
}

// ---------------------------------------------------------------------------
// K5: attention softmax over time + weighted pooling. One block per batch row.
// ---------------------------------------------------------------------------
__global__ void __launch_bounds__(256)
pool(const _Float16* __restrict__ HT, const float* __restrict__ attn_v,
     float* __restrict__ C) {
  const int b = blockIdx.x, tid = threadIdx.x;
  __shared__ float sc[Sn];
  __shared__ float red[Sn];

  {  // score[s] = (h_s . attn_v) / sqrt(H)
    const _Float16* p = HT + (tid * Bn + b) * Hn;
    float acc = 0.f;
    for (int n = 0; n < Hn; ++n) acc += (float)p[n] * attn_v[n];
    sc[tid] = acc * 0.04419417382415922f;  // 1/sqrt(512)
  }
  __syncthreads();
  red[tid] = sc[tid]; __syncthreads();
  for (int o = 128; o > 0; o >>= 1) {
    if (tid < o) red[tid] = fmaxf(red[tid], red[tid + o]);
    __syncthreads();
  }
  const float mx = red[0]; __syncthreads();
  const float e = __expf(sc[tid] - mx);
  sc[tid] = e; red[tid] = e; __syncthreads();
  for (int o = 128; o > 0; o >>= 1) {
    if (tid < o) red[tid] += red[tid + o];
    __syncthreads();
  }
  const float inv = 1.f / red[0];

  for (int n = tid; n < Hn; n += 256) {
    float acc = 0.f;
    for (int s = 0; s < Sn; ++s)
      acc += sc[s] * (float)HT[(s * Bn + b) * Hn + n];
    C[b * Hn + n] = acc * inv;
  }
}

// ---------------------------------------------------------------------------
extern "C" void kernel_launch(void* const* d_in, const int* in_sizes, int n_in,
                              void* d_out, int out_size, void* d_ws, size_t ws_size,
                              hipStream_t stream) {
  (void)in_sizes; (void)n_in; (void)out_size; (void)ws_size;
  const float* x    = (const float*)d_in[0];
  const float* t    = (const float*)d_in[1];
  const float* mask = (const float*)d_in[2];
  const float* w_gx = (const float*)d_in[3];
  const float* b_gx = (const float*)d_in[4];
  const float* W_gh = (const float*)d_in[5];
  const float* b_gh = (const float*)d_in[6];
  const float* Wz   = (const float*)d_in[7];
  const float* Uz   = (const float*)d_in[8];
  const float* Vz   = (const float*)d_in[9];
  const float* bz   = (const float*)d_in[10];
  const float* Wr   = (const float*)d_in[11];
  const float* Ur   = (const float*)d_in[12];
  const float* Vr   = (const float*)d_in[13];
  const float* br   = (const float*)d_in[14];
  const float* Wh   = (const float*)d_in[15];
  const float* Uh   = (const float*)d_in[16];
  const float* Vh   = (const float*)d_in[17];
  const float* bh   = (const float*)d_in[18];
  const float* attn_v = (const float*)d_in[19];
  float* C = (float*)d_out;

  char* ws = (char*)d_ws;
  size_t off = 0;
  auto take = [&](size_t bytes) -> char* {
    char* p = ws + off;
    off += (bytes + 255) & ~size_t(255);
    return p;
  };
  float*     x_mean = (float*)take(Dn * sizeof(float));
  _Float16*  WzT  = (_Float16*)take(Dn * Hn * 2);
  _Float16*  VzT  = (_Float16*)take(Dn * Hn * 2);
  _Float16*  WrT  = (_Float16*)take(Dn * Hn * 2);
  _Float16*  VrT  = (_Float16*)take(Dn * Hn * 2);
  _Float16*  WhT  = (_Float16*)take(Dn * Hn * 2);
  _Float16*  VhT  = (_Float16*)take(Dn * Hn * 2);
  _Float16*  WghT = (_Float16*)take(Dn * Hn * 2);
  _Float16*  UzT  = (_Float16*)take((size_t)Hn * Hn * 2);
  _Float16*  UrT  = (_Float16*)take((size_t)Hn * Hn * 2);
  _Float16*  UhT  = (_Float16*)take((size_t)Hn * Hn * 2);
  _Float16*  Xh   = (_Float16*)take((size_t)Mn * Dn * 2);
  _Float16*  Mh   = (_Float16*)take((size_t)Mn * Dn * 2);
  _Float16*  Dl   = (_Float16*)take((size_t)Mn * Dn * 2);
  _Float16*  AZ   = (_Float16*)take((size_t)Mn * Hn * 2);
  _Float16*  AR   = (_Float16*)take((size_t)Mn * Hn * 2);
  _Float16*  AH   = (_Float16*)take((size_t)Mn * Hn * 2);
  _Float16*  GH   = (_Float16*)take((size_t)Mn * Hn * 2);
  _Float16*  HT   = (_Float16*)take((size_t)Mn * Hn * 2);

  // K0: pack/transpose weights to f16 (N,K)
  {
    const int e1 = Dn * Hn, b1 = (e1 + 255) / 256;
    pack_w<<<b1, 256, 0, stream>>>(Wz,   WzT,  Dn, Hn);
    pack_w<<<b1, 256, 0, stream>>>(Vz,   VzT,  Dn, Hn);
    pack_w<<<b1, 256, 0, stream>>>(Wr,   WrT,  Dn, Hn);
    pack_w<<<b1, 256, 0, stream>>>(Vr,   VrT,  Dn, Hn);
    pack_w<<<b1, 256, 0, stream>>>(Wh,   WhT,  Dn, Hn);
    pack_w<<<b1, 256, 0, stream>>>(Vh,   VhT,  Dn, Hn);
    pack_w<<<b1, 256, 0, stream>>>(W_gh, WghT, Dn, Hn);
    const int e2 = Hn * Hn, b2 = (e2 + 255) / 256;
    pack_w<<<b2, 256, 0, stream>>>(Uz, UzT, Hn, Hn);
    pack_w<<<b2, 256, 0, stream>>>(Ur, UrT, Hn, Hn);
    pack_w<<<b2, 256, 0, stream>>>(Uh, UhT, Hn, Hn);
  }

  // K1: feature means
  mean_k<<<Dn, 256, 0, stream>>>(x, mask, x_mean);

  // K2: sequential per-sequence scan -> Xh, Mh, Dl (f16 time-major)
  scan_k<<<Bn, Dn, 0, stream>>>(x, t, mask, w_gx, b_gx, x_mean, Xh, Mh, Dl);

  // K3: fused input projections -> AZ, AR, AH, GH
  gemm1<<<dim3(Mn / 16, 8), 256, 0, stream>>>(Xh, Mh, Dl,
                                              WzT, VzT, WrT, VrT, WhT, VhT, WghT,
                                              bz, br, bh, b_gh,
                                              AZ, AR, AH, GH);

  // K4: persistent GRU recurrence -> HT
  recur<<<Bn / 16, 256, 0, stream>>>(AZ, AR, AH, GH, UzT, UrT, UhT, HT);

  // K5: attention pooling -> C
  pool<<<Bn, 256, 0, stream>>>(HT, attn_v, C);
}
